// PieceWiseLinear_39419209842749
// MI455X (gfx1250) — matrix operand order repaired and checked by reading, not compiled, not run on hardware
//
#include <hip/hip_runtime.h>
#include <hip/hip_bf16.h>

// PieceWiseLinear: out[b, f*64+e] = relu( sum_k enc[b,f,k]*W[f,k,e] + bias[f,e] )
// enc[b,f,k] = saturate((X[b,f]-lo[f,k]) / (hi[f,k]-lo[f,k]))
// B=32768, NF=64, NB=33 (K=32 bins), E=64.

typedef __attribute__((ext_vector_type(16))) _Float16 v16h;
typedef __attribute__((ext_vector_type(8)))  float    v8f;

#define PWL_NF 64
#define PWL_NB 33
#define PWL_NK 32
#define PWL_E  64
#define PWL_OUTW (PWL_NF * PWL_E)   // 4096 floats per output row

// One wave = one feature f, 8 M-tiles of 16 rows, all 4 N-tiles of 16 cols.
// v_wmma_f32_16x16x32_f16: K=32 == full reduction in a single WMMA.
__global__ __launch_bounds__(256) void
PieceWiseLinear_39419209842749_kernel(const float* __restrict__ X,
                                      const float* __restrict__ bounds,
                                      const float* __restrict__ W,
                                      const float* __restrict__ bias,
                                      float* __restrict__ out,
                                      int wgPerF)   // (B/128) wave-groups per feature
{
    const int lane  = threadIdx.x & 31;
    const int wave  = threadIdx.x >> 5;
    const int wId   = blockIdx.x * 8 + wave;

    const int f      = wId / wgPerF;          // feature, 0..63
    const int mgroup = wId % wgPerF;          // which chunk of 128 rows
    const int rowBase = mgroup * 128;

    const int half = lane >> 4;               // 0: lanes 0-15, 1: lanes 16-31
    const int ln   = lane & 15;

    // ---- per-wave constants -------------------------------------------------
    // A-operand K mapping (16-bit A 16x32): lane<16 -> K in {0..7, 16..23},
    // lane>=16 -> K in {8..15, 24..31}; element h of v16h:
    //   kA(h) = (half?8:0) + (h<8 ? h : h+8)
    const float* __restrict__ bf = bounds + f * PWL_NB;
    float lo[16], rs[16];
#pragma unroll
    for (int h = 0; h < 16; ++h) {
        const int k = (half ? 8 : 0) + (h < 8 ? h : h + 8);
        const float l  = bf[k];
        const float hi = bf[k + 1];
        lo[h] = l;
        rs[h] = 1.0f / (hi - l);
    }

    // B-operand (W[f] as 32x16 tiles): same K mapping as A:
    // element h of v16h -> K = (half?8:0) + (h<8 ? h : h+8), N = ln.
    v16h Bop[4];
#pragma unroll
    for (int nt = 0; nt < 4; ++nt) {
        const int n = nt * 16 + ln;
#pragma unroll
        for (int h = 0; h < 16; ++h) {
            const int k = (half ? 8 : 0) + (h < 8 ? h : h + 8);
            Bop[nt][h] = (_Float16)W[(f * PWL_NK + k) * PWL_E + n];
        }
    }

    float bb[4];
#pragma unroll
    for (int nt = 0; nt < 4; ++nt)
        bb[nt] = bias[f * PWL_E + nt * 16 + ln];

    // ---- main loop over 8 M-tiles of 16 rows --------------------------------
    for (int t = 0; t < 8; ++t) {
        const int mrow = rowBase + t * 16 + ln;    // this lane's A row
        const float x  = X[(size_t)mrow * PWL_NF + f];

        // Build A operand: enc = saturate((x - lo) * rs), cast to f16.
        v16h A;
#pragma unroll
        for (int h = 0; h < 16; ++h) {
            float e = (x - lo[h]) * rs[h];
            e = fminf(fmaxf(e, 0.0f), 1.0f);
            A[h] = (_Float16)e;
        }

        // 4 WMMAs: one per 16-col N-tile, full K=32 reduction each.
        v8f acc[4];
#pragma unroll
        for (int nt = 0; nt < 4; ++nt) {
            v8f c = {};
            acc[nt] = __builtin_amdgcn_wmma_f32_16x16x32_f16(
                /*neg_a=*/false, A, /*neg_b=*/false, Bop[nt],
                /*c_mod=*/(short)0, c, /*reuse_a=*/false, /*reuse_b=*/false);
        }

        // C/D layout: VGPR r, lanes 0-15 -> M=r, lanes 16-31 -> M=8+r; N=ln.
        const int rowD = rowBase + t * 16 + half * 8;
#pragma unroll
        for (int nt = 0; nt < 4; ++nt) {
            float* __restrict__ o =
                out + (size_t)rowD * PWL_OUTW + f * PWL_E + nt * 16 + ln;
#pragma unroll
            for (int r = 0; r < 8; ++r) {
                float v = acc[nt][r] + bb[nt];
                v = v > 0.0f ? v : 0.0f;
                o[(size_t)r * PWL_OUTW] = v;   // 64B-contiguous per half-wave
            }
        }
    }
}

extern "C" void kernel_launch(void* const* d_in, const int* in_sizes, int n_in,
                              void* d_out, int out_size, void* d_ws, size_t ws_size,
                              hipStream_t stream) {
    const float* X      = (const float*)d_in[0];   // (B, 64)
    const float* bounds = (const float*)d_in[1];   // (64, 33)
    const float* W      = (const float*)d_in[2];   // (64, 32, 64)
    const float* bias   = (const float*)d_in[3];   // (64, 64)
    float*       out    = (float*)d_out;           // (B, 4096)

    const int B = in_sizes[0] / PWL_NF;            // 32768
    const int wgPerF = B / 128;                    // 256 wave-groups per feature
    const int totalWaves = PWL_NF * wgPerF;        // 16384
    const int blocks = totalWaves / 8;             // 2048 blocks of 8 waves

    PieceWiseLinear_39419209842749_kernel<<<blocks, 256, 0, stream>>>(
        X, bounds, W, bias, out, wgPerF);
}
